// PaddedEncoderRNN_33956011442326
// MI455X (gfx1250) — compile-verified, matrix operand order checked by reading, not run on hardware
//
#include <hip/hip_runtime.h>

typedef __bf16 bf16;
typedef __attribute__((ext_vector_type(16))) __bf16 bf16x16;
typedef __attribute__((ext_vector_type(8)))  float  floatx8;

#define T_SEQ  512
#define BATCH  64
#define EDIM   512
#define G3     (3 * EDIM)   // 1536
#define NLAYER 2
#define LDSW   520          // padded LDS row stride for W slice (bank spread)
#define HPW    196          // padded row stride for hp staging tile
#define NWG_REC 8

// ---------------------------------------------------------------------------
// WMMA 16x16x32 bf16 fragment loader (row-major source, ld in elements).
// ISA 7.12.2 16-bit A layout: lanes 0-15 hold row r0+lane, K {k0..k0+7, k0+16..k0+23};
// lanes 16-31 hold same rows, K {k0+8..k0+15, k0+24..k0+31}. B (K x N, stored as
// W[n][k] row-major) uses the mirrored layout, so the same loader serves both.
// ---------------------------------------------------------------------------
__device__ inline bf16x16 load_frag16(const bf16* __restrict__ base, int ld, int r0, int k0) {
  int lane = threadIdx.x & 31;
  int half = lane >> 4;
  int row  = lane & 15;
  const bf16* p = base + (size_t)(r0 + row) * ld + k0 + half * 8;
  bf16x16 f;
#pragma unroll
  for (int e = 0; e < 8; ++e) f[e] = p[e];          // K = k0 + half*8 + 0..7
#pragma unroll
  for (int e = 0; e < 8; ++e) f[e + 8] = p[e + 16]; // K = k0 + half*8 + 16..23
  return f;
}

// ---------------------------------------------------------------------------
// Grid-wide sense barrier (8 co-resident WGs; 1 WG per WGP at 250KB LDS each).
// ---------------------------------------------------------------------------
__device__ inline void grid_barrier(unsigned int* cnt, unsigned int* gen, unsigned int nwg) {
  __syncthreads();
  if (threadIdx.x == 0) {
    __threadfence();
    unsigned int g = atomicAdd(gen, 0u);
    unsigned int arrived = atomicAdd(cnt, 1u);
    if (arrived == nwg - 1u) {
      atomicExch(cnt, 0u);
      __threadfence();
      atomicAdd(gen, 1u);
    } else {
      while (atomicAdd(gen, 0u) == g) { __builtin_amdgcn_s_sleep(2); }
    }
    __threadfence();
  }
  __syncthreads();
}

// ---------------------------------------------------------------------------
__global__ void init_barrier_kernel(unsigned int* bar) {
  if (threadIdx.x == 0) { bar[0] = 0u; bar[1] = 0u; }
}

__global__ void convert_weights_kernel(const float* __restrict__ wih,
                                       const float* __restrict__ whh,
                                       bf16* __restrict__ wih_bf,
                                       bf16* __restrict__ whh_bf, int n) {
  int i = blockIdx.x * blockDim.x + threadIdx.x;
  if (i < n) {
    wih_bf[i] = (bf16)wih[i];
    whh_bf[i] = (bf16)whh[i];
  }
}

__global__ void embed_kernel(const int* __restrict__ tokens,
                             const float* __restrict__ emb,
                             bf16* __restrict__ xbf) {
  int i   = blockIdx.x * blockDim.x + threadIdx.x;   // over T*B*E
  int row = i >> 9;                                   // / EDIM
  int e   = i & (EDIM - 1);
  int tok = tokens[row];
  xbf[i] = (bf16)emb[(size_t)tok * EDIM + e];
}

// ---------------------------------------------------------------------------
// x_proj = xbf @ W_ih^T + b_ih   (M=T*B=32768, N=1536, K=512), bf16 WMMA.
// Each wave owns a 16x64 strip (4 accumulators) so the A fragment is loaded
// once per K-step and reused across 4 WMMAs. 8 waves/block span 512 N columns.
// ---------------------------------------------------------------------------
__global__ void __launch_bounds__(256)
xproj_kernel(const bf16* __restrict__ xbf,
             const bf16* __restrict__ wih_bf,  // layer base [G3, E] row-major
             const float* __restrict__ bih,    // layer base [G3]
             float* __restrict__ xproj) {      // [T*B, G3]
  int wave = threadIdx.x >> 5;
  int lane = threadIdx.x & 31;
  int m0 = blockIdx.x * 16;
  int nb = (blockIdx.y * 8 + wave) * 64;   // strip base column

  floatx8 acc[4];
#pragma unroll
  for (int j = 0; j < 4; ++j) acc[j] = (floatx8){};

  for (int k0 = 0; k0 < EDIM; k0 += 32) {
    bf16x16 a = load_frag16(xbf, EDIM, m0, k0);
#pragma unroll
    for (int j = 0; j < 4; ++j) {
      bf16x16 b = load_frag16(wih_bf, EDIM, nb + j * 16, k0);
      acc[j] = __builtin_amdgcn_wmma_f32_16x16x32_bf16(false, a, false, b, (short)0,
                                                       acc[j], false, false);
    }
  }
  int half = lane >> 4;
  int col  = lane & 15;
#pragma unroll
  for (int j = 0; j < 4; ++j) {
    int n0 = nb + j * 16;
    float bias = bih[n0 + col];
#pragma unroll
    for (int r = 0; r < 8; ++r) {
      int m = m0 + r + half * 8;
      xproj[(size_t)m * G3 + n0 + col] = acc[j][r] + bias;
    }
  }
}

// ---------------------------------------------------------------------------
// Persistent GRU recurrence. 8 WGs, WG owns h columns [c0, c0+64).
// LDS: W_hh slice [192][LDSW] bf16 (196KB) + hp staging [64][HPW] f32 (50KB)
//      + lens cache (256B).
// Per step: prefetch this step's x_proj gate segments (covered by the GEMM),
// hp = h @ Wslice^T via WMMA (each wave: fixed batch-row tile, 6 n-tiles,
// A fragment loaded once per K-step), gates, write own h slice, grid barrier.
// ---------------------------------------------------------------------------
__global__ void __launch_bounds__(256)
gru_recurrence_kernel(const bf16* __restrict__ whh_bf,   // layer base [G3, E]
                      const float* __restrict__ bhh,     // layer base [G3]
                      const float* __restrict__ xproj,   // [T*B, G3]
                      const int*  __restrict__ lens,     // [B]
                      bf16* __restrict__ xbf,            // [T*B, E] -> y (next layer in)
                      bf16* __restrict__ hbuf,           // [2, B, E] double buffer
                      float* __restrict__ out_x,         // [T,B,E] (last layer only)
                      float* __restrict__ out_hidden,    // layer base [B, E]
                      unsigned int* __restrict__ bar,
                      int write_out) {
  extern __shared__ char smem_raw[];
  bf16*  Wsh   = (bf16*)smem_raw;                                   // [192][LDSW]
  float* hpsh  = (float*)(smem_raw + 192 * LDSW * sizeof(bf16));    // [64][HPW]
  int*   lensh = (int*)(smem_raw + 192 * LDSW * sizeof(bf16)
                                 + BATCH * HPW * sizeof(float));    // [64]

  const int tid  = threadIdx.x;
  const int wave = tid >> 5;
  const int lane = tid & 31;
  const int half = lane >> 4;
  const int colL = lane & 15;
  const int c0   = blockIdx.x * 64;

  const int mt = wave & 3;    // fixed batch-row tile for this wave
  const int ng = wave >> 2;   // n-tile group: tiles ng*6 .. ng*6+5

  // Stage this WG's 192 W_hh rows (gates r,z,n for columns c0..c0+63) into LDS.
  for (int idx = tid; idx < 192 * EDIM; idx += 256) {
    int j = idx >> 9;             // 0..191
    int k = idx & (EDIM - 1);
    int g = (j >> 6) * EDIM + c0 + (j & 63);   // proj row index in [0, 3E)
    Wsh[j * LDSW + k] = whh_bf[(size_t)g * EDIM + k];
  }
  if (tid < BATCH) lensh[tid] = lens[tid];
  // h0 = 0 (buffer 0, own slice)
  for (int idx = tid; idx < BATCH * 64; idx += 256) {
    int b = idx >> 6, j = idx & 63;
    hbuf[(size_t)b * EDIM + c0 + j] = (bf16)0.0f;
  }
  grid_barrier(&bar[0], &bar[1], gridDim.x);

  // Per-thread loop-invariant gate constants: column j = tid & 63 is fixed.
  const int jcol = tid & 63;
  const int colT = c0 + jcol;
  const float bhr = bhh[colT];
  const float bhz = bhh[EDIM + colT];
  const float bhn = bhh[2 * EDIM + colT];

  // Prefetch mapping: threads 0..191 cover 64 batch rows x 3 gate segments of
  // 64 floats (256B) each in this WG's x_proj slice.
  const int pfb = tid & 63;     // batch row
  const int pfs = tid >> 6;     // gate segment 0..3 (3 is idle)

  for (int t = 0; t < T_SEQ; ++t) {
    const bf16* hOld = hbuf + (size_t)(t & 1) * BATCH * EDIM;
    bf16*       hNew = hbuf + (size_t)((t + 1) & 1) * BATCH * EDIM;
    const float* xp  = xproj + (size_t)t * BATCH * G3;

    // Warm this step's gate inputs into cache while the GEMM runs
    // (emits global_prefetch_b8; one-touch stream, never naturally cache-hot).
    if (pfs < 3) {
      const float* pf = xp + (size_t)pfb * G3 + pfs * EDIM + c0;
      __builtin_prefetch(pf, 0, 3);
      __builtin_prefetch(pf + 32, 0, 3);
    }

    // hp[64 x 192] = hOld[64 x 512] @ Wslice^T
    floatx8 acc[6];
#pragma unroll
    for (int i = 0; i < 6; ++i) acc[i] = (floatx8){};
    for (int k0 = 0; k0 < EDIM; k0 += 32) {
      bf16x16 a = load_frag16(hOld, EDIM, mt * 16, k0);
#pragma unroll
      for (int i = 0; i < 6; ++i) {
        bf16x16 b = load_frag16(Wsh, LDSW, (ng * 6 + i) * 16, k0);
        acc[i] = __builtin_amdgcn_wmma_f32_16x16x32_bf16(false, a, false, b, (short)0,
                                                         acc[i], false, false);
      }
    }
#pragma unroll
    for (int i = 0; i < 6; ++i) {
      int nt = ng * 6 + i;
#pragma unroll
      for (int r = 0; r < 8; ++r)
        hpsh[(mt * 16 + r + half * 8) * HPW + nt * 16 + colL] = acc[i][r];
    }
    __syncthreads();

    // Gate math for own 64x64 slice of h.
    for (int idx = tid; idx < BATCH * 64; idx += 256) {
      int b = idx >> 6;           // jcol is idx & 63 == tid & 63 (stride 256)
      size_t rb = (size_t)b * G3;
      float xr = xp[rb + colT];
      float xz = xp[rb + EDIM + colT];
      float xn = xp[rb + 2 * EDIM + colT];
      float hr = hpsh[b * HPW + jcol]       + bhr;
      float hz = hpsh[b * HPW + 64 + jcol]  + bhz;
      float hn = hpsh[b * HPW + 128 + jcol] + bhn;
      float r = 1.0f / (1.0f + __expf(-(xr + hr)));
      float z = 1.0f / (1.0f + __expf(-(xz + hz)));
      float n = tanhf(xn + r * hn);
      float hp = (float)hOld[(size_t)b * EDIM + colT];
      bool  m  = t < lensh[b];
      float hnew = m ? ((1.0f - z) * n + z * hp) : hp;   // hold past seq end
      float y    = m ? hnew : 0.0f;                      // zero-fill padding
      hNew[(size_t)b * EDIM + colT] = (bf16)hnew;
      size_t orow = ((size_t)t * BATCH + b) * EDIM + colT;
      xbf[orow] = (bf16)y;                               // next layer's input
      if (write_out) out_x[orow] = y;
    }
    grid_barrier(&bar[0], &bar[1], gridDim.x);
  }

  // Final hidden state for this layer (T even -> buffer 0).
  const bf16* hFin = hbuf + (size_t)(T_SEQ & 1) * BATCH * EDIM;
  for (int idx = tid; idx < BATCH * 64; idx += 256) {
    int b = idx >> 6, j = idx & 63;
    out_hidden[(size_t)b * EDIM + c0 + j] = (float)hFin[(size_t)b * EDIM + c0 + j];
  }
}

// ---------------------------------------------------------------------------
extern "C" void kernel_launch(void* const* d_in, const int* in_sizes, int n_in,
                              void* d_out, int out_size, void* d_ws, size_t ws_size,
                              hipStream_t stream) {
  const int*   tokens = (const int*)d_in[0];
  const int*   lens   = (const int*)d_in[1];
  const float* emb    = (const float*)d_in[2];
  const float* W_ih   = (const float*)d_in[3];
  const float* W_hh   = (const float*)d_in[4];
  const float* b_ih   = (const float*)d_in[5];
  const float* b_hh   = (const float*)d_in[6];

  float* out   = (float*)d_out;
  float* out_x = out;                                     // [T,B,E]
  float* out_h = out + (size_t)T_SEQ * BATCH * EDIM;      // [L,B,E]

  char* ws = (char*)d_ws;
  size_t off = 0;
  bf16* xbf = (bf16*)(ws + off);     off += (size_t)T_SEQ * BATCH * EDIM * sizeof(bf16);
  float* xproj = (float*)(ws + off); off += (size_t)T_SEQ * BATCH * G3 * sizeof(float);
  bf16* wih_bf = (bf16*)(ws + off);  off += (size_t)NLAYER * G3 * EDIM * sizeof(bf16);
  bf16* whh_bf = (bf16*)(ws + off);  off += (size_t)NLAYER * G3 * EDIM * sizeof(bf16);
  bf16* hbuf = (bf16*)(ws + off);    off += (size_t)2 * BATCH * EDIM * sizeof(bf16);
  unsigned int* bar = (unsigned int*)(ws + off);

  init_barrier_kernel<<<1, 64, 0, stream>>>(bar);

  {
    int n = NLAYER * G3 * EDIM;
    convert_weights_kernel<<<(n + 255) / 256, 256, 0, stream>>>(W_ih, W_hh, wih_bf,
                                                                whh_bf, n);
  }
  embed_kernel<<<(T_SEQ * BATCH * EDIM) / 256, 256, 0, stream>>>(tokens, emb, xbf);

  const size_t lds_bytes = 192 * LDSW * sizeof(bf16) + BATCH * HPW * sizeof(float)
                         + BATCH * sizeof(int);
  for (int l = 0; l < NLAYER; ++l) {
    dim3 grid(T_SEQ * BATCH / 16, G3 / (16 * 4 * 8));   // 2048 x 3
    xproj_kernel<<<grid, 256, 0, stream>>>(xbf, wih_bf + (size_t)l * G3 * EDIM,
                                           b_ih + l * G3, xproj);
    gru_recurrence_kernel<<<NWG_REC, 256, lds_bytes, stream>>>(
        whh_bf + (size_t)l * G3 * EDIM, b_hh + l * G3, xproj, lens, xbf, hbuf,
        out_x, out_h + (size_t)l * BATCH * EDIM, bar,
        (l == NLAYER - 1) ? 1 : 0);
  }
}
